// GaussianPriorAttention_39745627357554
// MI455X (gfx1250) — matrix operand order repaired and checked by reading, not compiled, not run on hardware
//
#include <hip/hip_runtime.h>
#include <hip/hip_bf16.h>
#include <cmath>

// ---------------------------------------------------------------------------
// GaussianPriorAttention on MI455X (gfx1250, wave32, WMMA 16x16x32 f16->f32)
//
//   1) f32->f16 convert (vectorized x4) of x, Wq, Wk, Wv, Wo
//   2) WMMA GEMM (block 128x128, 8 waves, wave tile 32x64):
//        Q = (Xh @ Wqh^T + bq) * 0.125  -> [B,H,L,hd] f16   (scale folded in)
//        K =  Xh @ Wkh^T + bk           -> [B,H,L,hd] f16
//        V =  Xh @ Wvh^T + bv           -> [B,H,hd,L] f16   (transposed)
//   3) prior table tbl[d] = lambda*(0.8 e^{-(d-1)^2/0.5} + 0.4 e^{-(d-30)^2/8})
//   4) flash attention, 64 keys/iter, band-skip prior (zero for d>48),
//      row reductions via DPP16 butterflies (no ds_bpermute)
//   5) WMMA GEMM  out = Ctx @ Woh^T + bo  (f32)
// ---------------------------------------------------------------------------

#define DM    1024
#define NH    16
#define HDV   64
#define BB    2
#define LL    2048
#define MTOT  (BB*LL)          // 4096

typedef __attribute__((ext_vector_type(16))) _Float16 v16h;
typedef __attribute__((ext_vector_type(8)))  float    v8f;
typedef __attribute__((ext_vector_type(8)))  _Float16 h8;
typedef __attribute__((ext_vector_type(4)))  float    f4;
typedef __attribute__((ext_vector_type(4)))  _Float16 h4;

union V16U { v16h v; h8 h[2]; };

#define WMMA_F16(a, b, c) \
  __builtin_amdgcn_wmma_f32_16x16x32_f16(false, (a), false, (b), (short)0, (c), false, false)

// DPP16 lane-exchange within a 16-lane row (VALU, no LDS pipe).
// 0xB1 = quad_perm(1,0,3,2) [xor1], 0x4E = quad_perm(2,3,0,1) [xor2],
// 0x141 = row_half_mirror [xor7], 0x140 = row_mirror [xor15].
template <int CTRL>
__device__ __forceinline__ float dppf(float x) {
  return __int_as_float(
      __builtin_amdgcn_update_dpp(0, __float_as_int(x), CTRL, 0xF, 0xF, true));
}

// A fragment: 16x32 f16 (M x K) from row-major [M,K], leading dim ld.
__device__ __forceinline__ v16h load_a_frag(const _Float16* base, int ld, int lane) {
  int r  = lane & 15;
  int kb = (lane >> 4) << 3;
  V16U u;
  u.h[0] = *(const h8*)(base + r * ld + kb);
  u.h[1] = *(const h8*)(base + r * ld + kb + 16);
  return u.v;
}

// B fragment: 32x16 f16 (K x N) from row-major [N,K] (i.e. B^T), leading dim ld.
__device__ __forceinline__ v16h load_b_frag(const _Float16* base, int ld, int lane) {
  int n  = lane & 15;
  int kb = (lane >> 4) << 3;
  V16U u;
  u.h[0] = *(const h8*)(base + n * ld + kb);
  u.h[1] = *(const h8*)(base + n * ld + kb + 16);
  return u.v;
}

// ---------------------------------------------------------------------------
__global__ void cvt_f32_f16_x4(const float* __restrict__ src,
                               _Float16* __restrict__ dst, int n4) {
  int i = blockIdx.x * blockDim.x + threadIdx.x;
  if (i < n4) {
    f4 s = *(const f4*)(src + 4 * (size_t)i);
    *(h4*)(dst + 4 * (size_t)i) = __builtin_convertvector(s, h4);
  }
}

__global__ void build_mask(const float* __restrict__ lam, float* __restrict__ tbl) {
  int i = blockIdx.x * blockDim.x + threadIdx.x;
  if (i < LL) {
    float d = (float)i;
    float m = 0.8f * __expf(-((d - 1.0f)  * (d - 1.0f))  * (1.0f / 0.5f))
            + 0.4f * __expf(-((d - 30.0f) * (d - 30.0f)) * (1.0f / 8.0f));
    tbl[i] = lam[0] * m;
  }
}

// ---------------------------------------------------------------------------
// GEMM: C[M,N] = (A[M,K] @ Bt[N,K]^T + bias[N]) * oscale
// 256 threads = 8 waves (4x2); block tile 128x128; wave tile 32x64 (2x4 frags).
// MODE 0: f32 row-major. MODE 1: f16 scatter [B,H,L,hd]. MODE 2: f16 scatter [B,H,hd,L].
template <int MODE>
__global__ __launch_bounds__(256) void gemm_nt(const _Float16* __restrict__ A,
                                               const _Float16* __restrict__ Bt,
                                               const float* __restrict__ bias,
                                               float oscale,
                                               float* __restrict__ outF,
                                               _Float16* __restrict__ outH,
                                               int M, int N, int K) {
  int lane = threadIdx.x & 31;
  int wave = threadIdx.x >> 5;
  int row0 = blockIdx.x * 128 + (wave >> 1) * 32;
  int col0 = blockIdx.y * 128 + (wave & 1) * 64;

  v8f acc[2][4] = {};

  const _Float16* Ar0 = A  + (size_t)row0 * K;
  const _Float16* Ar1 = A  + (size_t)(row0 + 16) * K;
  const _Float16* Bc  = Bt + (size_t)col0 * K;

  for (int k0 = 0; k0 < K; k0 += 32) {
    v16h a0 = load_a_frag(Ar0 + k0, K, lane);
    v16h a1 = load_a_frag(Ar1 + k0, K, lane);
#pragma unroll
    for (int j = 0; j < 4; ++j) {
      v16h b = load_b_frag(Bc + (size_t)(j * 16) * K + k0, K, lane);
      acc[0][j] = WMMA_F16(a0, b, acc[0][j]);
      acc[1][j] = WMMA_F16(a1, b, acc[1][j]);
    }
  }

  int mo = (lane >> 4) << 3;
  int nn = lane & 15;
#pragma unroll
  for (int i = 0; i < 2; ++i) {
#pragma unroll
    for (int j = 0; j < 4; ++j) {
#pragma unroll
      for (int v = 0; v < 8; ++v) {
        int m = row0 + i * 16 + mo + v;
        int n = col0 + j * 16 + nn;
        float val = (acc[i][j][v] + bias[n]) * oscale;
        if (MODE == 0) {
          outF[(size_t)m * N + n] = val;
        } else {
          int b = m >> 11, l = m & (LL - 1);
          int h = n >> 6,  d = n & (HDV - 1);
          if (MODE == 1)      // [B,H,L,hd]
            outH[(((size_t)(b * NH + h) * LL) + l) * HDV + d] = (_Float16)val;
          else                // [B,H,hd,L]  (V transposed)
            outH[(((size_t)(b * NH + h) * HDV) + d) * LL + l] = (_Float16)val;
        }
      }
    }
  }
}

// ---------------------------------------------------------------------------
// Flash attention with additive Gaussian prior (prior is zero for |q-k| > 48).
// grid = (L/64, B*H), block = 128 (4 waves, 16 queries per wave), 64 keys/iter.
__global__ __launch_bounds__(128) void attn_kernel(const _Float16* __restrict__ Qh,
                                                   const _Float16* __restrict__ Kh,
                                                   const _Float16* __restrict__ Vth,
                                                   const float* __restrict__ tbl,
                                                   _Float16* __restrict__ Ctx) {
  __shared__ float s_tbl[LL];
  __shared__ __align__(16) _Float16 s_p[4][16][72];  // per-wave 16x64 P tile (+pad)

  int tid  = threadIdx.x;
  int lane = tid & 31;
  int wave = tid >> 5;

  for (int i = tid; i < LL; i += 128) s_tbl[i] = tbl[i];
  __syncthreads();

  int bh = blockIdx.y;
  int b  = bh >> 4;
  int h  = bh & 15;
  int q0 = blockIdx.x * 64 + wave * 16;

  const _Float16* Qb = Qh  + ((size_t)bh * LL + q0) * HDV;
  const _Float16* Kb = Kh  + (size_t)bh * LL * HDV;
  const _Float16* Vb = Vth + (size_t)bh * HDV * LL;

  // Q fragments (already scaled by 1/sqrt(hd) in the projection)
  v16h qa0 = load_a_frag(Qb, HDV, lane);
  v16h qa1 = load_a_frag(Qb + 32, HDV, lane);

  v8f o[4] = {};
  float mrow[8], lrow[8];
#pragma unroll
  for (int v = 0; v < 8; ++v) { mrow[v] = -1e30f; lrow[v] = 0.0f; }

  int mo   = (lane >> 4) << 3;
  int ncol = lane & 15;
  int qrow = q0 + mo;

  for (int k0 = 0; k0 < LL; k0 += 64) {
    // ---- S tile: 16 queries x 64 keys (8 WMMA) ----
    v8f s[4] = {};
#pragma unroll
    for (int t = 0; t < 4; ++t) {
      const _Float16* kb_ = Kb + (size_t)(k0 + 16 * t) * HDV;
      s[t] = WMMA_F16(qa0, load_b_frag(kb_,      HDV, lane), s[t]);
      s[t] = WMMA_F16(qa1, load_b_frag(kb_ + 32, HDV, lane), s[t]);
    }

    // ---- additive Gaussian prior: only inside |q-k| <= 48 band (wave-uniform) ----
    float sc[4][8];
    int dist = (k0 > q0 + 15) ? (k0 - (q0 + 15))
             : ((q0 > k0 + 63) ? (q0 - (k0 + 63)) : 0);
    if (dist <= 48) {
#pragma unroll
      for (int t = 0; t < 4; ++t)
#pragma unroll
        for (int v = 0; v < 8; ++v) {
          int dd = (qrow + v) - (k0 + 16 * t + ncol);
          dd = dd < 0 ? -dd : dd;
          sc[t][v] = s[t][v] + s_tbl[dd];
        }
    } else {
#pragma unroll
      for (int t = 0; t < 4; ++t)
#pragma unroll
        for (int v = 0; v < 8; ++v) sc[t][v] = s[t][v];
    }

    // ---- online softmax; row reductions via DPP16 butterflies ----
    float mv[8];
#pragma unroll
    for (int v = 0; v < 8; ++v)
      mv[v] = fmaxf(fmaxf(sc[0][v], sc[1][v]), fmaxf(sc[2][v], sc[3][v]));
#pragma unroll
    for (int v = 0; v < 8; ++v) mv[v] = fmaxf(mv[v], dppf<0xB1>(mv[v]));   // xor1
#pragma unroll
    for (int v = 0; v < 8; ++v) mv[v] = fmaxf(mv[v], dppf<0x4E>(mv[v]));   // xor2
#pragma unroll
    for (int v = 0; v < 8; ++v) mv[v] = fmaxf(mv[v], dppf<0x141>(mv[v]));  // xor7
#pragma unroll
    for (int v = 0; v < 8; ++v) mv[v] = fmaxf(mv[v], dppf<0x140>(mv[v]));  // xor15

    float corr[8];
#pragma unroll
    for (int v = 0; v < 8; ++v) {
      float mn = fmaxf(mrow[v], mv[v]);
      corr[v]  = __expf(mrow[v] - mn);
      mrow[v]  = mn;
    }
#pragma unroll
    for (int t = 0; t < 4; ++t)
#pragma unroll
      for (int v = 0; v < 8; ++v) sc[t][v] = __expf(sc[t][v] - mrow[v]);

    float rs[8];
#pragma unroll
    for (int v = 0; v < 8; ++v)
      rs[v] = (sc[0][v] + sc[1][v]) + (sc[2][v] + sc[3][v]);
#pragma unroll
    for (int v = 0; v < 8; ++v) rs[v] += dppf<0xB1>(rs[v]);
#pragma unroll
    for (int v = 0; v < 8; ++v) rs[v] += dppf<0x4E>(rs[v]);
#pragma unroll
    for (int v = 0; v < 8; ++v) rs[v] += dppf<0x141>(rs[v]);
#pragma unroll
    for (int v = 0; v < 8; ++v) rs[v] += dppf<0x140>(rs[v]);

#pragma unroll
    for (int v = 0; v < 8; ++v) lrow[v] = lrow[v] * corr[v] + rs[v];
#pragma unroll
    for (int t = 0; t < 4; ++t)
#pragma unroll
      for (int v = 0; v < 8; ++v) o[t][v] *= corr[v];

    // ---- P (C-layout) -> LDS -> A-layout fragments ----
#pragma unroll
    for (int t = 0; t < 4; ++t)
#pragma unroll
      for (int v = 0; v < 8; ++v)
        s_p[wave][mo + v][ncol + 16 * t] = (_Float16)sc[t][v];
    asm volatile("" ::: "memory");  // DS ops are in-order within a wave
    v16h pa0 = load_a_frag(&s_p[wave][0][0],      72, lane);  // keys k0+0..31
    v16h pa1 = load_a_frag(&s_p[wave][0][0] + 32, 72, lane);  // keys k0+32..63

    // ---- O += P @ V  (8 WMMA; B frags from V^T rows, contiguous in k) ----
#pragma unroll
    for (int nt = 0; nt < 4; ++nt) {
      const _Float16* vb_ = Vb + (size_t)(nt * 16) * LL + k0;
      o[nt] = WMMA_F16(pa0, load_b_frag(vb_,      LL, lane), o[nt]);
      o[nt] = WMMA_F16(pa1, load_b_frag(vb_ + 32, LL, lane), o[nt]);
    }
  }

  // ---- normalize and store Ctx f16 as [B*L, D], columns h*64 + d ----
#pragma unroll
  for (int v = 0; v < 8; ++v) {
    float inv = 1.0f / lrow[v];
    int q = qrow + v;
    size_t off = ((size_t)b * LL + q) * DM + h * HDV;
#pragma unroll
    for (int nt = 0; nt < 4; ++nt)
      Ctx[off + nt * 16 + ncol] = (_Float16)(o[nt][v] * inv);
  }
}

// ---------------------------------------------------------------------------
extern "C" void kernel_launch(void* const* d_in, const int* in_sizes, int n_in,
                              void* d_out, int out_size, void* d_ws, size_t ws_size,
                              hipStream_t stream) {
  const float* x   = (const float*)d_in[0];
  const float* Wq  = (const float*)d_in[1];
  const float* bq  = (const float*)d_in[2];
  const float* Wk  = (const float*)d_in[3];
  const float* bk  = (const float*)d_in[4];
  const float* Wv  = (const float*)d_in[5];
  const float* bv  = (const float*)d_in[6];
  const float* Wo  = (const float*)d_in[7];
  const float* bo  = (const float*)d_in[8];
  const float* lam = (const float*)d_in[9];

  const size_t XN = (size_t)MTOT * DM;   // 4,194,304 halves
  const size_t WN = (size_t)DM * DM;     // 1,048,576 halves

  _Float16* Xh  = (_Float16*)d_ws;
  _Float16* Wqh = Xh  + XN;
  _Float16* Wkh = Wqh + WN;
  _Float16* Wvh = Wkh + WN;
  _Float16* Woh = Wvh + WN;
  _Float16* Qh  = Woh + WN;
  _Float16* Kh  = Qh  + XN;
  _Float16* Vth = Kh  + XN;
  _Float16* Ctx = Vth + XN;
  float*    tbl = (float*)(Ctx + XN);

  build_mask<<<(LL + 255) / 256, 256, 0, stream>>>(lam, tbl);
  cvt_f32_f16_x4<<<(int)((XN / 4 + 255) / 256), 256, 0, stream>>>(x,  Xh,  (int)(XN / 4));
  cvt_f32_f16_x4<<<(int)((WN / 4 + 255) / 256), 256, 0, stream>>>(Wq, Wqh, (int)(WN / 4));
  cvt_f32_f16_x4<<<(int)((WN / 4 + 255) / 256), 256, 0, stream>>>(Wk, Wkh, (int)(WN / 4));
  cvt_f32_f16_x4<<<(int)((WN / 4 + 255) / 256), 256, 0, stream>>>(Wv, Wvh, (int)(WN / 4));
  cvt_f32_f16_x4<<<(int)((WN / 4 + 255) / 256), 256, 0, stream>>>(Wo, Woh, (int)(WN / 4));

  dim3 gg(MTOT / 128, DM / 128);  // (32, 8)
  gemm_nt<1><<<gg, 256, 0, stream>>>(Xh, Wqh, bq, 0.125f, nullptr, Qh,  MTOT, DM, DM);
  gemm_nt<1><<<gg, 256, 0, stream>>>(Xh, Wkh, bk, 1.0f,   nullptr, Kh,  MTOT, DM, DM);
  gemm_nt<2><<<gg, 256, 0, stream>>>(Xh, Wvh, bv, 1.0f,   nullptr, Vth, MTOT, DM, DM);

  attn_kernel<<<dim3(LL / 64, BB * NH), 128, 0, stream>>>(Qh, Kh, Vth, tbl, Ctx);

  gemm_nt<0><<<gg, 256, 0, stream>>>(Ctx, Woh, bo, 1.0f, (float*)d_out, nullptr, MTOT, DM, DM);
}
